// PointSetPooling_76587856822815
// MI455X (gfx1250) — matrix operand order repaired and checked by reading, not compiled, not run on hardware
//
#include <hip/hip_runtime.h>

typedef __attribute__((ext_vector_type(16))) __bf16 v16bf;
typedef __attribute__((ext_vector_type(8)))  float  v8f;

#define BM 128
#define BN 64
#define BK 32
#define SA 40   // LDS stride (bf16) for A tile: 80B rows -> 16B-aligned b128 ops, stride 20 words
#define SB 40   // LDS stride for B tile
#define CS 72   // LDS stride for C write-back tile (144B rows, 16B aligned)
#define CPAD 320
#define BN_EPS 1e-5f

union FragU { unsigned int u[8]; uint4 q[2]; v16bf v; };
union Pack16 { uint4 q[2]; unsigned int u[8]; __bf16 h[16]; };

// ---------------------------------------------------------------------------
// Generic fused layer: Y = relu(Xn @ W^T + b), Xn = a*X + c (per input column),
// plus per-output-column sum / sum-of-squares accumulation for batchnorm.
// X: M x CinPad (bf16 or f32 by template), W: Cout x Cin (f32), Y: M x CoutPad bf16.
// Block: 256 threads = 8 waves; tile BM=128 x BN=64, K-step 32.
// ---------------------------------------------------------------------------
template<bool INBF16>
__global__ void __launch_bounds__(256)
k_gemm(const void* __restrict__ xin,
       const float* __restrict__ aff,          // [0..CPAD)=a, [CPAD..2*CPAD)=c
       const float* __restrict__ W, const float* __restrict__ bias,
       __bf16* __restrict__ xout, float* __restrict__ stats,  // [c]=sum, [CPAD+c]=sumsq
       int M, int Cin, int CinPad, int Cout, int CoutPad)
{
    __shared__ __align__(16) char smem[BM * CS * 2];   // 18432B: staging phase then C tile
    __bf16* As = (__bf16*)smem;                        // BM*SA*2 = 10240B
    __bf16* Bs = (__bf16*)(smem + BM * SA * 2);        // BN*SB*2 =  5120B
    __bf16* Cs = (__bf16*)smem;                        // aliased after compute
    __shared__ float sSum[BN], sSq[BN];

    const int tid  = threadIdx.x;
    const int lane = tid & 31;
    const int wv   = tid >> 5;               // 8 waves, each owns 16 rows of M
    const int row0 = blockIdx.x * BM;
    const int n0   = blockIdx.y * BN;

    if (tid < BN) { sSum[tid] = 0.f; sSq[tid] = 0.f; }

    v8f acc[BN / 16];
    #pragma unroll
    for (int t = 0; t < BN / 16; ++t)
        #pragma unroll
        for (int r = 0; r < 8; ++r) acc[t][r] = 0.f;

    const int laneoff = (lane & 16) ? 8 : 0;   // A-frag K half select
    const int koff    = (lane & 16) ? 8 : 0;   // B-frag K half select
    const int mrow    = (wv << 4) + (lane & 15);

    // staging coordinates (hoisted)
    const int ar  = tid >> 1;
    const int acs = (tid & 1) << 4;            // 0 or 16
    const int bn  = tid >> 2;
    const int bks = (tid & 3) << 3;            // 0,8,16,24
    const int gn  = n0 + bn;
    const int gncl = (gn < Cout) ? gn : (Cout - 1);

    for (int k0 = 0; k0 < CinPad; k0 += BK) {
        __syncthreads();
        // ---- stage A tile (128x32): wide loads, affine, pack, 2x b128 LDS store ----
        {
            const size_t base = (size_t)(row0 + ar) * CinPad + k0 + acs;
            float xv[16];
            if constexpr (INBF16) {
                Pack16 u;
                const uint4* src = (const uint4*)((const __bf16*)xin + base);
                u.q[0] = src[0]; u.q[1] = src[1];
                if (k0 + BK < CinPad)
                    __builtin_prefetch((const __bf16*)xin + base + BK, 0, 1);
                #pragma unroll
                for (int j = 0; j < 16; ++j) xv[j] = (float)u.h[j];
            } else {
                union { float4 q[4]; float f[16]; } u;
                const float4* src = (const float4*)((const float*)xin + base);
                #pragma unroll
                for (int q = 0; q < 4; ++q) u.q[q] = src[q];
                if (k0 + BK < CinPad)
                    __builtin_prefetch((const float*)xin + base + BK, 0, 1);
                #pragma unroll
                for (int j = 0; j < 16; ++j) xv[j] = u.f[j];
            }
            union { float4 q[4]; float f[16]; } ua, uc;
            const float4* av = (const float4*)(aff + k0 + acs);
            const float4* cv = (const float4*)(aff + CPAD + k0 + acs);
            #pragma unroll
            for (int q = 0; q < 4; ++q) { ua.q[q] = av[q]; uc.q[q] = cv[q]; }
            Pack16 p;
            #pragma unroll
            for (int j = 0; j < 16; ++j)
                p.h[j] = (__bf16)(ua.f[j] * xv[j] + uc.f[j]);
            uint4* dstA = (uint4*)&As[ar * SA + acs];
            dstA[0] = p.q[0]; dstA[1] = p.q[1];
        }
        // ---- stage B tile (64 n x 32 k), branchless clamp, 1x b128 LDS store ----
        {
            union { uint4 q; __bf16 h[8]; } p;
            #pragma unroll
            for (int e = 0; e < 8; ++e) {
                const int gk   = k0 + bks + e;
                const int gkcl = (gk < Cin) ? gk : (Cin - 1);
                float wval = W[(size_t)gncl * Cin + gkcl];
                if (gn >= Cout || gk >= Cin) wval = 0.f;
                p.h[e] = (__bf16)wval;
            }
            *(uint4*)&Bs[bn * SB + bks] = p.q;
        }
        __syncthreads();

        // ---- A fragment: two contiguous 16B runs -> 2x ds_load_b128 ----
        FragU fa;
        fa.q[0] = *(const uint4*)&As[mrow * SA + laneoff];
        fa.q[1] = *(const uint4*)&As[mrow * SA + 16 + laneoff];
        // ---- 4 WMMAs: wave covers 16 rows x 64 cols; B frag = two 16B runs ----
        #pragma unroll
        for (int t = 0; t < BN / 16; ++t) {
            const int n = (t << 4) + (lane & 15);
            FragU fb;
            fb.q[0] = *(const uint4*)&Bs[n * SB + koff];
            fb.q[1] = *(const uint4*)&Bs[n * SB + koff + 16];
            acc[t] = __builtin_amdgcn_wmma_f32_16x16x32_bf16(
                         false, fa.v, false, fb.v, (short)0, acc[t], false, false);
        }
    }

    __syncthreads();   // last fragment reads done -> safe to alias As/Bs as Cs

    // ---- epilogue: bias + relu -> Cs (LDS) + hierarchical BN stats ----
    const int colLane = lane & 15;
    const int rbase   = (lane & 16) ? 8 : 0;
    #pragma unroll
    for (int t = 0; t < BN / 16; ++t) {
        const int cl = (t << 4) + colLane;         // local col in [0,BN)
        const int cg = n0 + cl;
        const float bb = (cg < Cout) ? bias[cg] : 0.f;
        float ps = 0.f, pq = 0.f;
        #pragma unroll
        for (int r = 0; r < 8; ++r) {
            const float y = fmaxf(acc[t][r] + bb, 0.f);   // pad cols: acc=0,bb=0 -> 0
            Cs[((wv << 4) + rbase + r) * CS + cl] = (__bf16)y;
            ps += y; pq += y * y;
        }
        ps += __shfl_xor(ps, 16);
        pq += __shfl_xor(pq, 16);
        if (lane < 16 && cg < Cout) {
            atomicAdd(&sSum[cl], ps);
            atomicAdd(&sSq [cl], pq);
        }
    }
    __syncthreads();

    // ---- coalesced write-back: each thread stores 32 contiguous bf16 (64B) ----
    {
        const int lrow = tid >> 1;
        const int ch   = (tid & 1) << 5;           // 0 or 32
        const uint4* srcc = (const uint4*)(Cs + lrow * CS + ch);
        uint4* dst = (uint4*)(xout + (size_t)(row0 + lrow) * CoutPad + n0 + ch);
        #pragma unroll
        for (int q = 0; q < 4; ++q) dst[q] = srcc[q];
    }
    if (tid < BN) {
        const int cg = n0 + tid;
        if (cg < Cout) {
            atomicAdd(&stats[cg], sSum[tid]);
            atomicAdd(&stats[CPAD + cg], sSq[tid]);
        }
    }
}

// ---------------------------------------------------------------------------
// Layer 1: x = relu((pc[i0]-pc[kp[i1]]) @ W1^T + b1), K=3 too thin for WMMA.
// ---------------------------------------------------------------------------
__global__ void __launch_bounds__(256)
k_layer1(const float* __restrict__ pc, const int* __restrict__ kpi,
         const int* __restrict__ si, const float* __restrict__ W,
         const float* __restrict__ bias, __bf16* __restrict__ xout,
         float* __restrict__ stats)
{
    __shared__ float sW[96], sB[32], sSum[32], sSq[32];
    const int tid = threadIdx.x;
    if (tid < 96) sW[tid] = W[tid];
    if (tid < 32) { sB[tid] = bias[tid]; sSum[tid] = 0.f; sSq[tid] = 0.f; }
    __syncthreads();

    const int s  = blockIdx.x * 256 + tid;
    const int i0 = si[2 * s], i1 = si[2 * s + 1];
    const int kp = kpi[i1];
    const float d0 = pc[3 * i0]     - pc[3 * kp];
    const float d1 = pc[3 * i0 + 1] - pc[3 * kp + 1];
    const float d2 = pc[3 * i0 + 2] - pc[3 * kp + 2];
    const int lane = tid & 31;

    Pack16 pk0, pk1;
    #pragma unroll
    for (int c = 0; c < 32; ++c) {
        float y = fmaf(sW[3*c], d0, fmaf(sW[3*c+1], d1, fmaf(sW[3*c+2], d2, sB[c])));
        y = fmaxf(y, 0.f);
        if (c < 16) pk0.h[c] = (__bf16)y; else pk1.h[c - 16] = (__bf16)y;
        float ps = y, pq = y * y;
        #pragma unroll
        for (int o = 16; o; o >>= 1) { ps += __shfl_xor(ps, o); pq += __shfl_xor(pq, o); }
        if (lane == 0) { atomicAdd(&sSum[c], ps); atomicAdd(&sSq[c], pq); }
    }
    uint4* dst = (uint4*)&xout[(size_t)s * 32];
    dst[0] = pk0.q[0]; dst[1] = pk0.q[1]; dst[2] = pk1.q[0]; dst[3] = pk1.q[1];
    __syncthreads();
    if (tid < 32) { atomicAdd(&stats[tid], sSum[tid]); atomicAdd(&stats[CPAD + tid], sSq[tid]); }
}

// stats (sum,sumsq) + (gamma,beta) -> per-column affine a*x + c
__global__ void k_affine(const float* __restrict__ stats, float* __restrict__ aff,
                         const float* __restrict__ g, const float* __restrict__ bt,
                         float invM, int Cout)
{
    const int c = threadIdx.x;            // blockDim = CPAD
    float a = 0.f, cc = 0.f;
    if (c < Cout) {
        const float mean = stats[c] * invM;
        const float var  = stats[CPAD + c] * invM - mean * mean;
        a  = g[c] * rsqrtf(var + BN_EPS);
        cc = bt[c] - mean * a;
    }
    aff[c] = a; aff[CPAD + c] = cc;
}

__global__ void k_identity(float* __restrict__ aff)
{
    const int c = threadIdx.x;
    aff[c] = 1.f; aff[CPAD + c] = 0.f;
}

// segment-max with layer-4 BN affine applied per element; clamp-at-0 folded in
// (all values >= 0 -> int-bitcast atomicMax is order-preserving; seg pre-zeroed).
__global__ void __launch_bounds__(256)
k_segmax(const __bf16* __restrict__ x4, const int* __restrict__ si,
         const float* __restrict__ aff, int* __restrict__ seg, long total)
{
    const long idx = (long)blockIdx.x * 256 + threadIdx.x;
    if (idx >= total) return;
    const int s = (int)(idx / 300);
    const int c = (int)(idx - (long)s * 300);
    const int segid = si[2 * s + 1];
    float v = aff[c] * (float)x4[(size_t)s * CPAD + c] + aff[CPAD + c];
    v = fmaxf(v, 0.f);
    atomicMax(seg + (size_t)segid * CPAD + c, __float_as_int(v));
}

// final BN affine of out-MLP layer 2 -> dense f32 output (K x 300, unpadded)
__global__ void __launch_bounds__(256)
k_final(const __bf16* __restrict__ x6, const float* __restrict__ aff,
        float* __restrict__ out, int total)
{
    const int idx = blockIdx.x * 256 + threadIdx.x;
    if (idx >= total) return;
    const int s = idx / 300, c = idx - s * 300;
    out[idx] = aff[c] * (float)x6[(size_t)s * CPAD + c] + aff[CPAD + c];
}

extern "C" void kernel_launch(void* const* d_in, const int* in_sizes, int n_in,
                              void* d_out, int out_size, void* d_ws, size_t ws_size,
                              hipStream_t stream)
{
    (void)in_sizes; (void)n_in; (void)out_size; (void)ws_size;
    const int S  = 1048576;
    const int Kp = 32768;

    const float* pc  = (const float*)d_in[0];
    const int*   kpi = (const int*)d_in[1];
    const int*   si  = (const int*)d_in[2];
    // point_params: 4 layers x (W,b,g,bt) at d_in[3..18]; out_params at d_in[19..26]
    const float* Wp[6]; const float* bp[6]; const float* gp[6]; const float* btp[6];
    for (int l = 0; l < 4; ++l) {
        Wp[l]  = (const float*)d_in[3 + 4*l];  bp[l]  = (const float*)d_in[4 + 4*l];
        gp[l]  = (const float*)d_in[5 + 4*l];  btp[l] = (const float*)d_in[6 + 4*l];
    }
    for (int l = 0; l < 2; ++l) {
        Wp[4+l]  = (const float*)d_in[19 + 4*l]; bp[4+l]  = (const float*)d_in[20 + 4*l];
        gp[4+l]  = (const float*)d_in[21 + 4*l]; btp[4+l] = (const float*)d_in[22 + 4*l];
    }

    // ---- workspace layout (aliased over the kernel timeline) ----
    char* ws = (char*)d_ws;
    size_t off = 0;
    float* stats = (float*)(ws + off); off += (size_t)6 * 2 * CPAD * 4;   // 15 KB
    float* affb  = (float*)(ws + off); off += (size_t)7 * 2 * CPAD * 4;   // slot0=identity
    off = (off + 255) & ~(size_t)255;
    __bf16* x3 = (__bf16*)(ws + off);  off += (size_t)S * 128 * 2;        // 256 MB (live into layer 4)
    char* dyn = ws + off;                                                 // aliased region
    __bf16* x1  = (__bf16*)dyn;                                           //  64 MB
    __bf16* x2  = (__bf16*)(dyn + (size_t)S * 32 * 2);                    // 128 MB
    __bf16* x4  = (__bf16*)dyn;                                           // 640 MB (x1,x2 dead)
    int*    seg = (int*)(dyn + (size_t)S * CPAD * 2);                     //  42 MB
    __bf16* x5  = (__bf16*)dyn;                                           //  21 MB (x4 dead)
    __bf16* x6  = (__bf16*)(dyn + ((size_t)32 << 20));                    //  21 MB

    const float invS = 1.f / (float)S, invK = 1.f / (float)Kp;
    float* st = stats; float* af = affb;                 // slot helpers
    #define STATS(l) (st + (size_t)(l) * 2 * CPAD)
    #define AFF(l)   (af + (size_t)(l) * 2 * CPAD)      // slot l (0 = identity)

    hipMemsetAsync(stats, 0, (size_t)6 * 2 * CPAD * 4, stream);
    hipMemsetAsync(seg,   0, (size_t)Kp * CPAD * 4, stream);
    k_identity<<<1, CPAD, 0, stream>>>(AFF(0));

    // point MLP
    k_layer1<<<S / 256, 256, 0, stream>>>(pc, kpi, si, Wp[0], bp[0], x1, STATS(0));
    k_affine<<<1, CPAD, 0, stream>>>(STATS(0), AFF(1), gp[0], btp[0], invS, 32);

    k_gemm<true><<<dim3(S / BM, 1), 256, 0, stream>>>(x1, AFF(1), Wp[1], bp[1], x2, STATS(1),
                                                      S, 32, 32, 64, 64);
    k_affine<<<1, CPAD, 0, stream>>>(STATS(1), AFF(2), gp[1], btp[1], invS, 64);

    k_gemm<true><<<dim3(S / BM, 2), 256, 0, stream>>>(x2, AFF(2), Wp[2], bp[2], x3, STATS(2),
                                                      S, 64, 64, 128, 128);
    k_affine<<<1, CPAD, 0, stream>>>(STATS(2), AFF(3), gp[2], btp[2], invS, 128);

    k_gemm<true><<<dim3(S / BM, 5), 256, 0, stream>>>(x3, AFF(3), Wp[3], bp[3], x4, STATS(3),
                                                      S, 128, 128, 300, CPAD);
    k_affine<<<1, CPAD, 0, stream>>>(STATS(3), AFF(4), gp[3], btp[3], invS, 300);

    // segment max (BN affine + clamp folded in)
    const long totS = (long)S * 300;
    k_segmax<<<(unsigned)((totS + 255) / 256), 256, 0, stream>>>(x4, si, AFF(4), seg, totS);

    // out MLP (seg buffer holds non-negative f32 bit patterns)
    k_gemm<false><<<dim3(Kp / BM, 5), 256, 0, stream>>>(seg, AFF(0), Wp[4], bp[4], x5, STATS(4),
                                                        Kp, 300, CPAD, 300, CPAD);
    k_affine<<<1, CPAD, 0, stream>>>(STATS(4), AFF(5), gp[4], btp[4], invK, 300);

    k_gemm<true><<<dim3(Kp / BM, 5), 256, 0, stream>>>(x5, AFF(5), Wp[5], bp[5], x6, STATS(5),
                                                       Kp, 300, CPAD, 300, CPAD);
    k_affine<<<1, CPAD, 0, stream>>>(STATS(5), AFF(6), gp[5], btp[5], invK, 300);

    const int totK = Kp * 300;
    k_final<<<(totK + 255) / 256, 256, 0, stream>>>(x6, AFF(6), (float*)d_out, totK);
    #undef STATS
    #undef AFF
}
